// ResBlock_41059887350294
// MI455X (gfx1250) — compile-verified, matrix operand order checked by reading, not run on hardware
//
#include <hip/hip_runtime.h>
#include <stdint.h>

// Problem constants (match reference)
#define Bsz    64
#define Edim   100000
#define Cdim   80000
#define Nnodes 10000
#define NNZv   1600000
#define LN_EPS 1e-5f

typedef __attribute__((ext_vector_type(4)))  unsigned int v4u;
typedef __attribute__((ext_vector_type(8)))  int          v8i;
typedef __attribute__((ext_vector_type(4)))  int          v4i;
typedef __attribute__((ext_vector_type(2)))  float        v2f;
typedef __attribute__((ext_vector_type(8)))  float        v8f;
typedef __attribute__((ext_vector_type(16))) _Float16     v16h;

#if defined(__has_builtin)
#if __has_builtin(__builtin_amdgcn_tensor_load_to_lds) && __has_builtin(__builtin_amdgcn_s_wait_tensorcnt)
#define HAVE_TDM 1
#endif
#if __has_builtin(__builtin_amdgcn_wmma_f32_16x16x4_f32)
#define HAVE_WMMA_F32X4 1
#endif
#endif
#ifndef HAVE_TDM
#define HAVE_TDM 0
#endif
#ifndef HAVE_WMMA_F32X4
#define HAVE_WMMA_F32X4 0
#endif

// ---------------------------------------------------------------------------
// TDM: issue a 1-D async copy of `tile_elems` dwords (n_valid real, OOB reads
// return 0 -> tail zero-fill) from global `gaddr` into LDS byte addr.
// D# per CDNA5 ISA ch.8: group0 {count=1, lds_addr, global_addr, type=2},
// group1 {data_size=4B, tensor_dim0=n_valid, tensor_dim1=1, tile_dim0=tile}.
// Tracked with TENSORcnt; waited via s_wait_tensorcnt.
// This toolchain exposes the 6-arg builtin:
//   (uint32x4 g0, int32x8 g1, int32x4 g2, int32x4 g3, int32x8 extra, i32 cpol)
// Groups 2/3 + extra stay zero: tile_dim1..4 == 0 marks them unused (1-D copy).
// ---------------------------------------------------------------------------
#if HAVE_TDM
__device__ __forceinline__ void tdm_load_1d(uint32_t lds_byte_addr, const void* gaddr,
                                            uint32_t n_valid, uint32_t tile_elems) {
  const uint64_t ga = (uint64_t)(uintptr_t)gaddr;
  v4u g0;
  g0.x = 1u;                                             // count=1, user desc
  g0.y = lds_byte_addr;                                  // lds_addr
  g0.z = (uint32_t)(ga & 0xFFFFFFFFu);                   // global_addr[31:0]
  g0.w = (uint32_t)((ga >> 32) & 0x01FFFFFFu) | (2u << 30); // ga[56:32] | type=2
  v8i g1;
  g1[0] = (int)(2u << 16);                               // data_size = 4 bytes
  g1[1] = (int)((n_valid & 0xFFFFu) << 16);              // tensor_dim0 lo16
  g1[2] = (int)(((n_valid >> 16) & 0xFFFFu) | (1u << 16)); // tensor_dim0 hi | tensor_dim1=1
  g1[3] = (int)((tile_elems & 0xFFFFu) << 16);           // tile_dim0
  g1[4] = 0;                                             // tile_dim1/2 unused
  g1[5] = (int)tile_elems;                               // tensor_dim0_stride
  g1[6] = 0;
  g1[7] = 0;
  v4i z4 = {0, 0, 0, 0};
  v8i z8 = {0, 0, 0, 0, 0, 0, 0, 0};
  __builtin_amdgcn_tensor_load_to_lds(g0, g1, z4, z4, z8, 0);
}
#endif

// ---------------------------------------------------------------------------
// xT[e*64 + b] = x[b*E + e]   (tiled LDS transpose, both sides coalesced)
// ---------------------------------------------------------------------------
__global__ void __launch_bounds__(256)
transpose_in_kernel(const float* __restrict__ x, float* __restrict__ xT) {
  __shared__ float tile[32][33];
  const int e0 = blockIdx.x * 32;
  const int b0 = blockIdx.y * 32;
  const int tx = threadIdx.x, ty = threadIdx.y;
  for (int i = ty; i < 32; i += 8)
    tile[i][tx] = x[(size_t)(b0 + i) * Edim + (e0 + tx)];
  __syncthreads();
  for (int i = ty; i < 32; i += 8)
    xT[(size_t)(e0 + i) * Bsz + (b0 + tx)] = tile[tx][i];
}

// dstT[r*64 + b] = bias[r]
__global__ void __launch_bounds__(256)
fill_bias_kernel(float* __restrict__ dstT, const float* __restrict__ bias, unsigned total) {
  const unsigned idx = blockIdx.x * blockDim.x + threadIdx.x;
  if (idx < total) dstT[idx] = bias[idx >> 6];
}

// ---------------------------------------------------------------------------
// SpMM scatter: dstT[cols[k]*64 + b] += vals[k] * srcT[rows[k]*64 + b]
// One wave handles one nonzero across all 64 batches (float2 per lane).
// Index stream (rows/cols/vals) staged into LDS by the Tensor Data Mover,
// double-buffered; compute waves overlap TDM of the next tile.
// ---------------------------------------------------------------------------
template <int TK>
__global__ void __launch_bounds__(256)
spmm_kernel(const float* __restrict__ srcT,
            const int* __restrict__ rows, const int* __restrict__ cols,
            const float* __restrict__ vals,
            float* __restrict__ dstT, int nnz) {
  __shared__ int   sh_rows[2][TK];
  __shared__ int   sh_cols[2][TK];
  __shared__ float sh_vals[2][TK];

  const int lane   = threadIdx.x & 31;
  const int wave   = threadIdx.x >> 5;
  const int nwaves = blockDim.x >> 5;
  const int ntiles = (nnz + TK - 1) / TK;

#if HAVE_TDM
  auto issue = [&](int tile, int buf) {
    const int start = tile * TK;
    const int rem   = nnz - start;
    const uint32_t nvalid = (uint32_t)(rem < TK ? rem : TK);
    tdm_load_1d((uint32_t)(uintptr_t)&sh_rows[buf][0], rows + start, nvalid, TK);
    tdm_load_1d((uint32_t)(uintptr_t)&sh_cols[buf][0], cols + start, nvalid, TK);
    tdm_load_1d((uint32_t)(uintptr_t)&sh_vals[buf][0], vals + start, nvalid, TK);
  };
  if ((int)blockIdx.x < ntiles && wave == 0) issue(blockIdx.x, 0);
#endif

  int it = 0;
  for (int t = blockIdx.x; t < ntiles; t += gridDim.x, ++it) {
    const int buf = it & 1;
#if HAVE_TDM
    if (wave == 0) __builtin_amdgcn_s_wait_tensorcnt(0);
    __syncthreads();                       // buf now valid for all waves
    const int tnext = t + gridDim.x;
    if (wave == 0 && tnext < ntiles) issue(tnext, buf ^ 1);  // prefetch next tile
#else
    const int start = t * TK;
    for (int i = threadIdx.x; i < TK; i += blockDim.x) {
      const int g = start + i;
      sh_rows[buf][i] = (g < nnz) ? rows[g] : 0;
      sh_cols[buf][i] = (g < nnz) ? cols[g] : 0;
      sh_vals[buf][i] = (g < nnz) ? vals[g] : 0.0f;
    }
    __syncthreads();
#endif
    const int per_wave = TK / nwaves;      // nonzeros per wave in this tile
    const int base = wave * per_wave;
    for (int i = 0; i < per_wave; ++i) {
      const int   r = sh_rows[buf][base + i];   // LDS broadcast reads
      const int   c = sh_cols[buf][base + i];
      const float v = sh_vals[buf][base + i];
      if (i + 1 < per_wave) {                   // prefetch next gather row
        const int rn = sh_rows[buf][base + i + 1];
        __builtin_prefetch((const void*)(srcT + (size_t)rn * Bsz + lane * 2), 0, 0);
      }
      const float2 xv = ((const float2*)(srcT + (size_t)r * Bsz))[lane]; // 256B/wave
      float* dst = dstT + (size_t)c * Bsz + lane * 2;
      unsafeAtomicAdd(dst + 0, v * xv.x);       // hw global_atomic_add_f32
      unsafeAtomicAdd(dst + 1, v * xv.y);
    }
    __syncthreads();                       // done with buf before TDM reuses it
  }
}

// ---------------------------------------------------------------------------
// GroupLayerNorm (8 contiguous channels per node) + ELU, in place on hT[C,64].
// Per-node column sums & sum-of-squares computed with WMMA: A = all-ones
// (layout-invariant), B = 4x16 / 32x16 tile of channel values, so every row of
// D holds the per-batch column sum. One wave per node; 4 chunks of 16 batches.
// ---------------------------------------------------------------------------
__global__ void __launch_bounds__(256)
groupnorm_elu_kernel(float* __restrict__ hT,
                     const float* __restrict__ gamma,
                     const float* __restrict__ beta) {
  const int lane = threadIdx.x & 31;
  const int wave = threadIdx.x >> 5;
  const int node = blockIdx.x * (blockDim.x >> 5) + wave;
  if (node >= Nnodes) return;              // wave-uniform (exact grid anyway)
  const int l15 = lane & 15;
  const int hi  = lane >> 4;
  const size_t ch0 = (size_t)node * 8;

  for (int chunk = 0; chunk < 4; ++chunk) {
    const int col = chunk * 16 + l15;
    float sum, sumsq;
#if HAVE_WMMA_F32X4
    v2f ones; ones.x = 1.0f; ones.y = 1.0f;
    const float a0 = hT[(ch0 + 0 + hi) * Bsz + col];
    const float a1 = hT[(ch0 + 2 + hi) * Bsz + col];
    const float a2 = hT[(ch0 + 4 + hi) * Bsz + col];
    const float a3 = hT[(ch0 + 6 + hi) * Bsz + col];
    v2f b0, b1, q0, q1;
    b0.x = a0;      b0.y = a1;      b1.x = a2;      b1.y = a3;
    q0.x = a0 * a0; q0.y = a1 * a1; q1.x = a2 * a2; q1.y = a3 * a3;
    v8f cs = {0.f,0.f,0.f,0.f,0.f,0.f,0.f,0.f};
    v8f cq = {0.f,0.f,0.f,0.f,0.f,0.f,0.f,0.f};
    cs = __builtin_amdgcn_wmma_f32_16x16x4_f32(false, ones, false, b0, (short)0, cs, false, false);
    cs = __builtin_amdgcn_wmma_f32_16x16x4_f32(false, ones, false, b1, (short)0, cs, false, false);
    cq = __builtin_amdgcn_wmma_f32_16x16x4_f32(false, ones, false, q0, (short)0, cq, false, false);
    cq = __builtin_amdgcn_wmma_f32_16x16x4_f32(false, ones, false, q1, (short)0, cq, false, false);
    sum = cs[0]; sumsq = cq[0];            // D row 0, N = lane&15 (documented C/D layout)
#else
    v16h onesA, bs, qs;
    for (int j = 0; j < 16; ++j) { onesA[j] = (_Float16)1.0f; bs[j] = (_Float16)0.0f; qs[j] = (_Float16)0.0f; }
    if (hi == 0) {                         // only one K-half carries data (no double count)
      for (int j = 0; j < 8; ++j) {
        const float a = hT[(ch0 + j) * Bsz + col];
        bs[j] = (_Float16)a;
        qs[j] = (_Float16)(a * a);
      }
    }
    v8f cs = {0.f,0.f,0.f,0.f,0.f,0.f,0.f,0.f};
    v8f cq = {0.f,0.f,0.f,0.f,0.f,0.f,0.f,0.f};
    cs = __builtin_amdgcn_wmma_f32_16x16x32_f16(false, onesA, false, bs, (short)0, cs, false, false);
    cq = __builtin_amdgcn_wmma_f32_16x16x32_f16(false, onesA, false, qs, (short)0, cq, false, false);
    sum = cs[0]; sumsq = cq[0];
#endif
    const float mean = sum * 0.125f;
    float var = sumsq * 0.125f - mean * mean;
    var = fmaxf(var, 0.0f);
    const float rs = rsqrtf(var + LN_EPS);
    // lanes 0-15 normalize channels 0..3, lanes 16-31 channels 4..7, batch=col
    for (int j = 0; j < 4; ++j) {
      const size_t ch = ch0 + (size_t)hi * 4 + j;
      const float v = hT[ch * Bsz + col];
      float y = gamma[ch] * ((v - mean) * rs) + beta[ch];
      y = (y > 0.0f) ? y : expm1f(y);      // ELU after affine, matches reference
      hT[ch * Bsz + col] = y;
    }
  }
}

// out[b*E + e] = outT[e*64 + b] + x[b*E + e]   (tiled transpose + residual)
__global__ void __launch_bounds__(256)
finish_kernel(const float* __restrict__ outT, const float* __restrict__ x,
              float* __restrict__ out) {
  __shared__ float tile[32][33];
  const int e0 = blockIdx.x * 32;
  const int b0 = blockIdx.y * 32;
  const int tx = threadIdx.x, ty = threadIdx.y;
  for (int i = ty; i < 32; i += 8)
    tile[i][tx] = outT[(size_t)(e0 + i) * Bsz + (b0 + tx)];
  __syncthreads();
  for (int i = ty; i < 32; i += 8) {
    const size_t o = (size_t)(b0 + i) * Edim + (e0 + tx);
    out[o] = tile[tx][i] + x[o];
  }
}

// ---------------------------------------------------------------------------
extern "C" void kernel_launch(void* const* d_in, const int* in_sizes, int n_in,
                              void* d_out, int out_size, void* d_ws, size_t ws_size,
                              hipStream_t stream) {
  (void)in_sizes; (void)n_in; (void)out_size; (void)ws_size;
  const float* x     = (const float*)d_in[0];
  const float* v_in  = (const float*)d_in[1];
  const float* b_in  = (const float*)d_in[2];
  const float* v_out = (const float*)d_in[3];
  const float* b_out = (const float*)d_in[4];
  const float* gamma = (const float*)d_in[5];
  const float* beta  = (const float*)d_in[6];
  const int* w_in_rows  = (const int*)d_in[7];
  const int* w_in_cols  = (const int*)d_in[8];
  const int* w_out_rows = (const int*)d_in[9];
  const int* w_out_cols = (const int*)d_in[10];
  // d_in[11] channel_groups: unused, groups are exactly c / 8 by construction
  float* out = (float*)d_out;

  // Workspace: xT [E,64] | hT [C,64] | outT [E,64]  (~72 MB, fits 192 MB L2)
  float* xT   = (float*)d_ws;
  float* hT   = xT + (size_t)Edim * Bsz;
  float* outT = hT + (size_t)Cdim * Bsz;

  const dim3 tb(32, 8);
  transpose_in_kernel<<<dim3(Edim / 32, Bsz / 32), tb, 0, stream>>>(x, xT);
  fill_bias_kernel<<<(Cdim * Bsz) / 256, 256, 0, stream>>>(hT, b_in, Cdim * Bsz);
  spmm_kernel<512><<<1024, 256, 0, stream>>>(xT, w_in_rows, w_in_cols, v_in, hT, NNZv);
  groupnorm_elu_kernel<<<Nnodes / 8, 256, 0, stream>>>(hT, gamma, beta);
  fill_bias_kernel<<<(Edim * Bsz) / 256, 256, 0, stream>>>(outT, b_out, Edim * Bsz);
  spmm_kernel<512><<<1024, 256, 0, stream>>>(hT, w_out_rows, w_out_cols, v_out, outT, NNZv);
  finish_kernel<<<dim3(Edim / 32, Bsz / 32), tb, 0, stream>>>(outT, x, out);
}
